// GraphUnet_910533067207
// MI455X (gfx1250) — matrix operand order repaired and checked by reading, not compiled, not run on hardware
//
#include <hip/hip_runtime.h>
#include <hip/hip_bf16.h>
#include <math.h>

// ---------------------------------------------------------------------------
// Graph U-Net forward for MI455X (gfx1250, wave32, WMMA).
// Heavy compute (13 convs + 20 GEMMs) via v_wmma_f32_16x16x32_f16:
// block tile 64M x 32N (each wave: 16x32 via 2 accumulators sharing one A
// fragment), fp32->f16 branchless LDS staging (packed b32 stores, 32-bit
// address selects), double-buffered pipeline, b128 LDS fragment loads,
// global_prefetch for K+2 tiles.
// ---------------------------------------------------------------------------

typedef __attribute__((ext_vector_type(16))) _Float16 v16h;
typedef __attribute__((ext_vector_type(8)))  _Float16 h8;
typedef __attribute__((ext_vector_type(2)))  _Float16 h2;
typedef __attribute__((ext_vector_type(8)))  float    v8f;

#define ALD 40   // A tile LDS row stride in halfs (80B rows -> 16B aligned)
#define BLD 40   // B tile LDS row stride in halfs ([n][k] transposed layout)

// A fragment 16x32 f16 (ISA layout): lanes 0-15 rows, K 0..7 & 16..23;
// lanes 16-31 same rows, K 8..15 & 24..31.  Two ds_load_b128 per lane.
__device__ inline v16h frag_a(const _Float16* As) {
  int lane = threadIdx.x & 31;
  int r  = lane & 15;
  int kb = (lane < 16) ? 0 : 8;
  const _Float16* p = As + r * ALD + kb;
  h8 lo = *(const h8*)(p);        // K = kb .. kb+7
  h8 hi = *(const h8*)(p + 16);   // K = kb+16 .. kb+23
  v16h a;
#pragma unroll
  for (int j = 0; j < 8; ++j) { a[j] = lo[j]; a[j + 8] = hi[j]; }
  return a;
}

// B fragment 32x16 f16: lane col c holds K 0..7 & 16..23 (lanes 0-15) or
// K 8..15 & 24..31 (lanes 16-31). LDS tile is stored transposed [n][k] so K is contiguous.
__device__ inline v16h frag_b(const _Float16* Bs) {
  int lane = threadIdx.x & 31;
  int c  = lane & 15;
  int kb = (lane < 16) ? 0 : 8;
  const _Float16* p = Bs + c * BLD + kb;
  h8 lo = *(const h8*)(p);        // K = kb .. kb+7
  h8 hi = *(const h8*)(p + 16);   // K = kb+16 .. kb+23
  v16h b;
#pragma unroll
  for (int j = 0; j < 8; ++j) { b[j] = lo[j]; b[j + 8] = hi[j]; }
  return b;
}

// ---------------------------------------------------------------------------
// Implicit-GEMM 3x3 SAME conv + fused BN + ReLU.
// 128 threads = 4 waves; block tile = 64 pixels x 32 Cout (2 wmma/wave/stage).
// kmeta[k] = { ci*H*W + dy*W + dx, (dy+1) | ((dx+1)<<8) }, preloaded to LDS.
// k in [K,Kp) has benign meta (dy=dx=0) and ZERO weights in wTn, so staging
// is branchless: spatial bounds -> 32-bit address/value selects only.
// ---------------------------------------------------------------------------
__global__ void __launch_bounds__(128)
k_conv3x3_wmma(const float* __restrict__ in, const _Float16* __restrict__ wTn,
               const int2* __restrict__ kmeta,
               const float* __restrict__ scale, const float* __restrict__ shift,
               float* __restrict__ out,
               int Cin, int H, int W, int Cout, int Kp) {
  __shared__ __align__(16) _Float16 As0[64 * ALD];
  __shared__ __align__(16) _Float16 As1[64 * ALD];
  __shared__ __align__(16) _Float16 Bs0[32 * BLD];
  __shared__ __align__(16) _Float16 Bs1[32 * BLD];
  __shared__ __align__(16) int2 kml[576];

  const int tid  = threadIdx.x;
  const int wave = tid >> 5;
  const int base = blockIdx.x * 64;
  const int HW = H * W;
  const int b   = base / HW;
  const int rem = base - b * HW;
  const int y   = rem / W;
  const int x0  = rem - y * W;
  const int n0  = blockIdx.y * 32;
  const int inB = b * Cin * HW + y * W + x0;   // all tensors < 2^31 elements

  for (int t = tid; t < Kp; t += 128) kml[t] = kmeta[t];
  __syncthreads();

  auto stageA = [&](int k0, _Float16* As) {
#pragma unroll
    for (int t = tid; t < 64 * 16; t += 128) {
      int kk2 = t >> 6;          // 0..15 (k pair)
      int m   = t & 63;          // consecutive lanes -> consecutive x
      int k   = k0 + kk2 * 2;
      int4 q = *(const int4*)(&kml[k]);   // meta for k, k+1 (one ds_load_b128)
      int dy0 = (q.y & 255) - 1, dx0 = (q.y >> 8) - 1;
      int dy1 = (q.w & 255) - 1, dx1 = (q.w >> 8) - 1;
      int yy0 = y + dy0, xx0 = x0 + m + dx0;
      int yy1 = y + dy1, xx1 = x0 + m + dx1;
      bool ok0 = (yy0 >= 0) & (yy0 < H) & (xx0 >= 0) & (xx0 < W);
      bool ok1 = (yy1 >= 0) & (yy1 < H) & (xx1 >= 0) & (xx1 < W);
      int o0 = ok0 ? (inB + q.x + m) : inB;   // 32-bit address select
      int o1 = ok1 ? (inB + q.z + m) : inB;
      float v0 = in[o0];
      float v1 = in[o1];
      v0 = ok0 ? v0 : 0.f;                    // value select
      v1 = ok1 ? v1 : 0.f;
      h2 p; p.x = (_Float16)v0; p.y = (_Float16)v1;
      *(h2*)(As + m * ALD + kk2 * 2) = p;     // packed ds_store_b32
    }
  };
  auto stageB = [&](int k0, _Float16* Bs) {
#pragma unroll
    for (int t = tid; t < 32 * 16; t += 128) {
      int n = t & 31, kk2 = t >> 5;
      h2 v; v.x = (_Float16)0.f; v.y = (_Float16)0.f;
      int n1 = n0 + n;
      if (n1 < Cout) v = *(const h2*)(wTn + (size_t)n1 * Kp + k0 + kk2 * 2);
      *(h2*)(Bs + n * BLD + kk2 * 2) = v;
    }
  };

  v8f acc0 = {}, acc1 = {};
  const int T = Kp >> 5;
  stageA(0, As0); stageB(0, Bs0);
  __syncthreads();
  for (int t = 0; t < T; t += 2) {
    if (t + 1 < T) { stageA((t + 1) * 32, As1); stageB((t + 1) * 32, Bs1); }
    {
      v16h a  = frag_a(As0 + wave * 16 * ALD);
      v16h b0 = frag_b(Bs0);
      v16h b1 = frag_b(Bs0 + 16 * BLD);
      acc0 = __builtin_amdgcn_wmma_f32_16x16x32_f16(false, a, false, b0,
                                                    (short)0, acc0, false, false);
      acc1 = __builtin_amdgcn_wmma_f32_16x16x32_f16(false, a, false, b1,
                                                    (short)0, acc1, false, false);
    }
    __syncthreads();
    if (t + 1 >= T) break;
    if (t + 2 < T) { stageA((t + 2) * 32, As0); stageB((t + 2) * 32, Bs0); }
    {
      v16h a  = frag_a(As1 + wave * 16 * ALD);
      v16h b0 = frag_b(Bs1);
      v16h b1 = frag_b(Bs1 + 16 * BLD);
      acc0 = __builtin_amdgcn_wmma_f32_16x16x32_f16(false, a, false, b0,
                                                    (short)0, acc0, false, false);
      acc1 = __builtin_amdgcn_wmma_f32_16x16x32_f16(false, a, false, b1,
                                                    (short)0, acc1, false, false);
    }
    __syncthreads();
  }

  // Epilogue: BN + ReLU; each lane's 8 results are x-contiguous -> 2x b128.
  int lane = tid & 31;
  int col  = lane & 15;
  int rb   = (lane < 16) ? 0 : 8;
  size_t px = (size_t)y * W + x0 + wave * 16 + rb;
#pragma unroll
  for (int half = 0; half < 2; ++half) {
    int n = n0 + half * 16 + col;
    v8f& acc = half ? acc1 : acc0;
    if (n < Cout) {
      float sc = scale[n], sh = shift[n];
      float4 o0, o1;
      o0.x = fmaxf(0.f, acc[0] * sc + sh);
      o0.y = fmaxf(0.f, acc[1] * sc + sh);
      o0.z = fmaxf(0.f, acc[2] * sc + sh);
      o0.w = fmaxf(0.f, acc[3] * sc + sh);
      o1.x = fmaxf(0.f, acc[4] * sc + sh);
      o1.y = fmaxf(0.f, acc[5] * sc + sh);
      o1.z = fmaxf(0.f, acc[6] * sc + sh);
      o1.w = fmaxf(0.f, acc[7] * sc + sh);
      float* dst = out + (size_t)(b * Cout + n) * HW + px;
      *(float4*)dst = o0;
      *(float4*)(dst + 4) = o1;
    }
  }
}

// ---------------------------------------------------------------------------
// Generic batched WMMA GEMM: C[z] = act(A[z] (MxK) @ B[z] (KxN) + bias).
// fp32 in/out, f16 compute.  REQUIRES: M%64==0, N%32==0, K%32==0
// (holds for every call site in this net) -> fully branchless staging.
// ---------------------------------------------------------------------------
__global__ void __launch_bounds__(128)
k_gemm_wmma(const float* __restrict__ A, const float* __restrict__ Bm,
            const float* __restrict__ bias, float* __restrict__ C,
            int M, int N, int K,
            long long sA, long long sB, long long sC, int relu) {
  __shared__ __align__(16) _Float16 As0[64 * ALD];
  __shared__ __align__(16) _Float16 As1[64 * ALD];
  __shared__ __align__(16) _Float16 Bs0[32 * BLD];
  __shared__ __align__(16) _Float16 Bs1[32 * BLD];
  (void)M;

  const int tid  = threadIdx.x;
  const int wave = tid >> 5;
  const int m0 = blockIdx.x * 64;
  const int n0 = blockIdx.y * 32;
  const long long z = blockIdx.z;
  const float* Ab = A + z * sA;
  const float* Bb = Bm + z * sB;
  float* Cb = C + z * sC;

  auto stageA = [&](int k0, _Float16* As) {
#pragma unroll
    for (int t = tid; t < 64 * 16; t += 128) {
      int m   = t >> 4;          // consecutive lanes -> consecutive k pairs
      int kk2 = t & 15;
      const float* src = Ab + (size_t)(m0 + m) * K + k0 + kk2 * 2;
      float2 v = *(const float2*)src;
      if (kk2 == 0 && k0 + 64 < K)
        __builtin_prefetch(src + 64, 0, 0);  // K+2 tile (global_prefetch_b8)
      h2 p; p.x = (_Float16)v.x; p.y = (_Float16)v.y;
      *(h2*)(As + m * ALD + kk2 * 2) = p;
    }
  };
  auto stageB = [&](int k0, _Float16* Bs) {
#pragma unroll
    for (int t = tid; t < 32 * 16; t += 128) {
      int n = t & 31, kk2 = t >> 5;
      int nn = n0 + n, k = k0 + kk2 * 2;
      float v0 = Bb[(size_t)k * N + nn];
      float v1 = Bb[(size_t)(k + 1) * N + nn];
      h2 p; p.x = (_Float16)v0; p.y = (_Float16)v1;
      *(h2*)(Bs + n * BLD + kk2 * 2) = p;
    }
  };

  v8f acc0 = {}, acc1 = {};
  const int T = K >> 5;
  stageA(0, As0); stageB(0, Bs0);
  __syncthreads();
  for (int t = 0; t < T; t += 2) {
    if (t + 1 < T) { stageA((t + 1) * 32, As1); stageB((t + 1) * 32, Bs1); }
    {
      v16h a  = frag_a(As0 + wave * 16 * ALD);
      v16h b0 = frag_b(Bs0);
      v16h b1 = frag_b(Bs0 + 16 * BLD);
      acc0 = __builtin_amdgcn_wmma_f32_16x16x32_f16(false, a, false, b0,
                                                    (short)0, acc0, false, false);
      acc1 = __builtin_amdgcn_wmma_f32_16x16x32_f16(false, a, false, b1,
                                                    (short)0, acc1, false, false);
    }
    __syncthreads();
    if (t + 1 >= T) break;
    if (t + 2 < T) { stageA((t + 2) * 32, As0); stageB((t + 2) * 32, Bs0); }
    {
      v16h a  = frag_a(As1 + wave * 16 * ALD);
      v16h b0 = frag_b(Bs1);
      v16h b1 = frag_b(Bs1 + 16 * BLD);
      acc0 = __builtin_amdgcn_wmma_f32_16x16x32_f16(false, a, false, b0,
                                                    (short)0, acc0, false, false);
      acc1 = __builtin_amdgcn_wmma_f32_16x16x32_f16(false, a, false, b1,
                                                    (short)0, acc1, false, false);
    }
    __syncthreads();
  }

  int lane = tid & 31;
  int col  = lane & 15;
  int rb   = (lane < 16) ? 0 : 8;
#pragma unroll
  for (int half = 0; half < 2; ++half) {
    int n = n0 + half * 16 + col;
    v8f& acc = half ? acc1 : acc0;
    float bv = bias ? bias[n] : 0.f;
#pragma unroll
    for (int r = 0; r < 8; ++r) {
      int m = m0 + wave * 16 + rb + r;
      float v = acc[r] + bv;
      if (relu) v = fmaxf(v, 0.f);
      Cb[(size_t)m * N + n] = v;
    }
  }
}

// --- conv prep: w (Cout,Cin,3,3) -> f16 [Cout][Kp] (k-major, zero padded);
//     BN folded to scale/shift; per-k patch metadata (padded region gets
//     benign dy=dx=0 meta; its weights are zero). ------------------------
__global__ void k_prep_conv(const float* __restrict__ w, const float* __restrict__ b,
                            const float* __restrict__ gamma, const float* __restrict__ beta,
                            const float* __restrict__ rm, const float* __restrict__ rv,
                            _Float16* __restrict__ wTn, int2* __restrict__ kmeta,
                            float* __restrict__ scale, float* __restrict__ shift,
                            int Cin, int Cout, int H, int W, int Kp) {
  int K = Cin * 9;
  int t = blockIdx.x * blockDim.x + threadIdx.x;
  if (t < Cout) {
    float sc = gamma[t] * rsqrtf(rv[t] + 1e-5f);
    scale[t] = sc;
    shift[t] = (b[t] - rm[t]) * sc + beta[t];
  }
  if (t < Kp) {
    if (t < K) {
      int ci = t / 9;
      int r9 = t - ci * 9;
      int dy = r9 / 3 - 1;
      int dx = r9 - (r9 / 3) * 3 - 1;
      kmeta[t] = make_int2(ci * H * W + dy * W + dx, (dy + 1) | ((dx + 1) << 8));
    } else {
      kmeta[t] = make_int2(0, 1 | (1 << 8));   // dy=dx=0, safe address
    }
  }
  int stride = gridDim.x * blockDim.x;
  for (int i = t; i < Cout * Kp; i += stride) {
    int n = i / Kp, k = i - n * Kp;
    float v = (k < K) ? w[(size_t)n * K + k] : 0.f;
    wTn[i] = (_Float16)v;
  }
}

// --- maxpool s=2 along W with flat argmax index ---------------------------
__global__ void k_maxpool2(const float* __restrict__ in, float* __restrict__ out,
                           int* __restrict__ idx, int Bn, int C, int H, int W) {
  int Wp = W >> 1;
  long long total = (long long)Bn * C * H * Wp;
  long long t = (long long)blockIdx.x * blockDim.x + threadIdx.x;
  if (t >= total) return;
  int wp = (int)(t % Wp); long long r = t / Wp;
  int y = (int)(r % H);   long long bc = r / H;
  const float* p = in + ((size_t)bc * H + y) * W + wp * 2;
  float v0 = p[0], v1 = p[1];
  int a = (v1 > v0) ? 1 : 0;
  out[t] = a ? v1 : v0;
  idx[t] = a + wp * 2 + y * W;
}

// --- gpool scores ---------------------------------------------------------
__global__ void k_rowdot(const float* __restrict__ h, const float* __restrict__ w,
                         const float* __restrict__ b, float* __restrict__ out,
                         int rows, int F) {
  int t = blockIdx.x * blockDim.x + threadIdx.x;
  if (t >= rows) return;
  const float* p = h + (size_t)t * F;
  float s = 0.f;
  for (int f = 0; f < F; ++f) s += p[f] * w[f];
  out[t] = s + b[0];
}

__global__ void k_scores(const float* __restrict__ wsc, float* __restrict__ sc,
                         int Bn, int C, int N) {
  int t = blockIdx.x * blockDim.x + threadIdx.x;
  if (t >= Bn * N) return;
  int b = t / N, n = t - b * N;
  float s = 0.f;
  for (int c = 0; c < C; ++c) s += wsc[((size_t)b * C + c) * N + n];
  s /= (float)C;
  sc[t] = 1.f / (1.f + expf(-s));
}

// --- top-k (N<=256), sorted desc, lowest index wins ties ------------------
__global__ void __launch_bounds__(256)
k_topk(const float* __restrict__ sc, float* __restrict__ vals,
       int* __restrict__ idx, int N, int nk) {
  __shared__ float s[256];
  __shared__ float rv[256];
  __shared__ int   ri[256];
  int b = blockIdx.x;
  int t = threadIdx.x;
  s[t] = (t < N) ? sc[b * N + t] : -1e30f;
  __syncthreads();
  for (int j = 0; j < nk; ++j) {
    rv[t] = s[t]; ri[t] = t;
    __syncthreads();
    for (int st = 128; st > 0; st >>= 1) {
      if (t < st) {
        float ov = rv[t + st]; int oi = ri[t + st];
        if (ov > rv[t] || (ov == rv[t] && oi < ri[t])) { rv[t] = ov; ri[t] = oi; }
      }
      __syncthreads();
    }
    if (t == 0) {
      vals[b * nk + j] = rv[0];
      idx[b * nk + j]  = ri[0];
      s[ri[0]] = -1e30f;
    }
    __syncthreads();
  }
}

__global__ void k_gather_scale(const float* __restrict__ h, const float* __restrict__ vals,
                               const int* __restrict__ idx, float* __restrict__ out,
                               int Bn, int C, int N, int nk, int F) {
  long long total = (long long)Bn * C * nk * F;
  long long t = (long long)blockIdx.x * blockDim.x + threadIdx.x;
  if (t >= total) return;
  int f = (int)(t % F); long long r = t / F;
  int i = (int)(r % nk); r /= nk;
  int c = (int)(r % C);  int b = (int)(r / C);
  out[t] = h[(((size_t)b * C + c) * N + idx[b * nk + i]) * F + f] * vals[b * nk + i];
}

// --- adjacency coarsening: (gb@gb != 0)[idx][:,idx] -----------------------
__global__ void k_coarsen(const float* __restrict__ g, const int* __restrict__ idx,
                          float* __restrict__ ung, int Bn, int N, int nk) {
  int t = blockIdx.x * blockDim.x + threadIdx.x;
  if (t >= Bn * nk * nk) return;
  int j = t % nk; int r = t / nk;
  int i = r % nk; int b = r / nk;
  int ri_ = idx[b * nk + i], cj = idx[b * nk + j];
  const float* grow = g + ((size_t)b * N + ri_) * N;
  const float* gcol = g + (size_t)b * N * N + cj;
  float v = 0.f;
  for (int k = 0; k < N; ++k)
    if (grow[k] != 0.f && gcol[(size_t)k * N] != 0.f) { v = 1.f; break; }
  ung[t] = v;
}

__global__ void k_colsum(const float* __restrict__ ung, float* __restrict__ cs,
                         int Bn, int nk) {
  int t = blockIdx.x * blockDim.x + threadIdx.x;
  if (t >= Bn * nk) return;
  int j = t % nk, b = t / nk;
  float s = 0.f;
  for (int i = 0; i < nk; ++i) s += ung[((size_t)b * nk + i) * nk + j];
  cs[t] = s;
}

__global__ void k_divcol(float* __restrict__ ung, const float* __restrict__ cs,
                         int Bn, int nk) {
  int t = blockIdx.x * blockDim.x + threadIdx.x;
  if (t >= Bn * nk * nk) return;
  int j = t % nk; int b = t / (nk * nk);
  ung[t] /= cs[b * nk + j];
}

// --- broadcast (C,N,F) -> (B,C,N,F) ---------------------------------------
__global__ void k_bcast(const float* __restrict__ src, float* __restrict__ dst,
                        long long n, int Bn) {
  long long t = (long long)blockIdx.x * blockDim.x + threadIdx.x;
  if (t >= n * Bn) return;
  dst[t] = src[t % n];
}

// --- graph unpool: inverse index + scatter-add with skip ------------------
__global__ void k_build_inv(const int* __restrict__ idx, int* __restrict__ inv,
                            int N, int nk) {
  int b = blockIdx.x;
  for (int n = threadIdx.x; n < N; n += blockDim.x) inv[b * N + n] = -1;
  __syncthreads();
  for (int i = threadIdx.x; i < nk; i += blockDim.x) inv[b * N + idx[b * nk + i]] = i;
}

__global__ void k_gunpool(const float* __restrict__ hin, const float* __restrict__ pre,
                          const int* __restrict__ inv, float* __restrict__ out,
                          int Bn, int C, int N, int nk, int F) {
  long long total = (long long)Bn * C * N * F;
  long long t = (long long)blockIdx.x * blockDim.x + threadIdx.x;
  if (t >= total) return;
  int f = (int)(t % F); long long r = t / F;
  int n = (int)(r % N); r /= N;
  int c = (int)(r % C); int b = (int)(r / C);
  int iv = inv[b * N + n];
  float v = pre[t];
  if (iv >= 0) v += hin[(((size_t)b * C + c) * nk + iv) * F + f];
  out[t] = v;
}

// --- max-unpool (s=2) fused with channel concat ---------------------------
__global__ void k_unpool_concat(const float* __restrict__ pooled, const int* __restrict__ find,
                                const float* __restrict__ dco, float* __restrict__ out,
                                int Bn, int C, int H, int W) {
  int Wp = W >> 1;
  long long total = (long long)Bn * 2 * C * H * W;
  long long t = (long long)blockIdx.x * blockDim.x + threadIdx.x;
  if (t >= total) return;
  int w = (int)(t % W); long long r = t / W;
  int y = (int)(r % H); r /= H;
  int c2 = (int)(r % (2 * C)); int b = (int)(r / (2 * C));
  float v;
  if (c2 < C) {
    long long pi = (((long long)b * C + c2) * H + y) * Wp + (w >> 1);
    v = (find[pi] == y * W + w) ? pooled[pi] : 0.f;
  } else {
    v = dco[(((long long)b * C + (c2 - C)) * H + y) * W + w];
  }
  out[t] = v;
}

__global__ void k_copyf(const float* __restrict__ s, float* __restrict__ d, long long n) {
  long long t = (long long)blockIdx.x * blockDim.x + threadIdx.x;
  if (t < n) d[t] = s[t];
}

__global__ void k_i2f(const int* __restrict__ s, float* __restrict__ d, int n) {
  int t = blockIdx.x * blockDim.x + threadIdx.x;
  if (t < n) d[t] = (float)s[t];
}

// ===========================================================================
// Host orchestration
// ===========================================================================
static inline size_t align256(size_t x) { return (x + 255) & ~((size_t)255); }

extern "C" void kernel_launch(void* const* d_in, const int* in_sizes, int n_in,
                              void* d_out, int out_size, void* d_ws, size_t ws_size,
                              hipStream_t stream) {
  (void)in_sizes; (void)n_in; (void)out_size; (void)ws_size;
  constexpr int B = 4;

  // --- input leaf indices (jax pytree flatten: dict keys sorted) ----------
  const float* g_in = (const float*)d_in[0];
  const float* h_in = (const float*)d_in[1];
  const int BOT1 = 2, BOT2 = 8, BOTG = 14;              // bottom_cnn1/2, bottom_gcn
  const int DC0A = 18, DC0B = 24, DC1A = 30, DC1B = 36; // down_cnns[0..3]
  const int DG0 = 42, DG1 = 46;                         // down_gcns
  const int GP0 = 50, GP1 = 52;                         // gpool_projs
  const int LAST = 54;                                  // last_cnn
  const int UPC0 = 60, UPC1 = 66, UPC2 = 72, UPC3 = 78, UPC4 = 84, UPC5 = 90;
  const int UG0 = 96, UG1 = 100;                        // up_gcns

  // --- workspace bump allocator -------------------------------------------
  char* ws = (char*)d_ws;
  size_t off = 0;
  auto allocb = [&](size_t bytes) { void* p = ws + off; off += align256(bytes); return p; };
  auto allocf = [&](size_t n) { return (float*)allocb(n * 4); };
  auto alloci = [&](size_t n) { return (int*)allocb(n * 4); };

  float* P0   = allocf(33554432);      // 4*32*256*1024 ping
  float* P1   = allocf(33554432);      // pong
  float* dco0 = allocf(16777216);      // (4,16,256,1024)
  float* dco1 = allocf(8388608);       // (4,32,128,512)
  float* dgo0 = allocf(8388608);       // (4,16,256,512)
  float* dgo1 = allocf(4194304);       // (4,32,128,256)
  int*   find0 = alloci(8388608);      // (4,16,256,512)
  int*   find1 = alloci(4194304);      // (4,32,128,256)
  float* g1   = allocf(65536);         // (4,128,128)
  float* g2b  = allocf(16384);         // (4,64,64)
  int*   gind0 = alloci(512);
  int*   gind1 = alloci(256);
  float* gvals0 = allocf(512);
  float* gvals1 = allocf(256);
  int*   inv0 = alloci(1024);
  int*   inv1 = alloci(512);
  float* S_hs  = allocf(2097152);
  float* S_t1  = allocf(1048576);
  float* S_hs2 = allocf(1048576);
  float* S_go  = allocf(2097152);
  float* S_wsc = allocf(16384);
  float* S_sc  = allocf(1024);
  float* S_cs  = allocf(512);
  _Float16* wTn = (_Float16*)allocb(40960 * 2);  // max Cout*Kp = 64*576
  int2* kmeta   = (int2*)allocb(1024 * 8);
  float* wscale = allocf(64);
  float* wshift = allocf(64);

  // --- helpers -------------------------------------------------------------
  auto conv = [&](const float* in, float* out, int base, int Cin, int Cout, int H, int W) {
    int K = Cin * 9;
    int Kp = (K + 31) & ~31;
    int tot = Cout * Kp;
    k_prep_conv<<<dim3((tot + 255) / 256), dim3(256), 0, stream>>>(
        (const float*)d_in[base + 5], (const float*)d_in[base + 0],
        (const float*)d_in[base + 2], (const float*)d_in[base + 1],
        (const float*)d_in[base + 3], (const float*)d_in[base + 4],
        wTn, kmeta, wscale, wshift, Cin, Cout, H, W, Kp);
    long long Mtot = (long long)B * H * W;
    dim3 grid((unsigned)(Mtot / 64), (unsigned)((Cout + 31) / 32), 1);
    k_conv3x3_wmma<<<grid, dim3(128), 0, stream>>>(in, wTn, kmeta, wscale, wshift,
                                                   out, Cin, H, W, Cout, Kp);
  };

  auto gemm = [&](const float* A, const float* Bm, const float* bias, float* C,
                  int M, int N, int K, long long sA, long long sB, long long sC,
                  int batch, int relu) {
    dim3 grid((unsigned)((M + 63) / 64), (unsigned)((N + 31) / 32), (unsigned)batch);
    k_gemm_wmma<<<grid, dim3(128), 0, stream>>>(A, Bm, bias, C, M, N, K, sA, sB, sC, relu);
  };

  // GCN: uses batch-0 of h, result broadcast across batch (matches reference).
  auto gcn = [&](const float* gadj, const float* h0, float* outB,
                 int C, int N, int F, int dh, int pbase) {
    const float* b1 = (const float*)d_in[pbase + 0];
    const float* b2 = (const float*)d_in[pbase + 1];
    const float* w1 = (const float*)d_in[pbase + 2];
    const float* w2 = (const float*)d_in[pbase + 3];
    gemm(gadj, h0, nullptr, S_hs, N, F, N, 0, (long long)N * F, (long long)N * F, C, 0);
    gemm(S_hs, w1, b1, S_t1, C * N, dh, F, 0, 0, 0, 1, 1);
    gemm(gadj, S_t1, nullptr, S_hs2, N, dh, N, 0, (long long)N * dh, (long long)N * dh, C, 0);
    gemm(S_hs2, w2, b2, S_go, C * N, F, dh, 0, 0, 0, 1, 1);
    long long n = (long long)C * N * F;
    k_bcast<<<dim3((unsigned)((n * B + 255) / 256)), dim3(256), 0, stream>>>(S_go, outB, n, B);
  };

  auto gpool = [&](const float* gadj, const float* hB, int C, int N, int F, int nk,
                   int pbase, float* gout, float* hout, float* vals, int* idx) {
    const float* pb = (const float*)d_in[pbase + 0];
    const float* pw = (const float*)d_in[pbase + 1];
    int rows = B * C * N;
    k_rowdot<<<dim3((rows + 255) / 256), dim3(256), 0, stream>>>(hB, pw, pb, S_wsc, rows, F);
    k_scores<<<dim3((B * N + 255) / 256), dim3(256), 0, stream>>>(S_wsc, S_sc, B, C, N);
    k_topk<<<dim3(B), dim3(256), 0, stream>>>(S_sc, vals, idx, N, nk);
    long long tot = (long long)B * C * nk * F;
    k_gather_scale<<<dim3((unsigned)((tot + 255) / 256)), dim3(256), 0, stream>>>(
        hB, vals, idx, hout, B, C, N, nk, F);
    k_coarsen<<<dim3((B * nk * nk + 255) / 256), dim3(256), 0, stream>>>(gadj, idx, gout, B, N, nk);
    k_colsum<<<dim3((B * nk + 255) / 256), dim3(256), 0, stream>>>(gout, S_cs, B, nk);
    k_divcol<<<dim3((B * nk * nk + 255) / 256), dim3(256), 0, stream>>>(gout, S_cs, B, nk);
  };

  auto gunpool = [&](const float* hin, const float* pre, const int* gidx,
                     int C, int N, int nk, int F, int* inv, float* out) {
    k_build_inv<<<dim3(B), dim3(256), 0, stream>>>(gidx, inv, N, nk);
    long long tot = (long long)B * C * N * F;
    k_gunpool<<<dim3((unsigned)((tot + 255) / 256)), dim3(256), 0, stream>>>(
        hin, pre, inv, out, B, C, N, nk, F);
  };

  auto unpool_concat = [&](const float* pooled, const int* find, const float* dco,
                           float* out, int C, int H, int W) {
    long long tot = (long long)B * 2 * C * H * W;
    k_unpool_concat<<<dim3((unsigned)((tot + 255) / 256)), dim3(256), 0, stream>>>(
        pooled, find, dco, out, B, C, H, W);
  };

  // ============================ DOWN PATH ================================
  conv(h_in, P0, DC0A, 1, 16, 256, 1024);                 // (4,16,256,1024)
  conv(P0, dco0, DC0B, 16, 16, 256, 1024);                // skip dco0
  {
    long long tot = (long long)B * 16 * 256 * 512;
    k_maxpool2<<<dim3((unsigned)((tot + 255) / 256)), dim3(256), 0, stream>>>(
        dco0, P0, find0, B, 16, 256, 1024);               // (4,16,256,512)
  }
  gcn(g_in, P0, dgo0, 16, 256, 512, 256, DG0);            // dgo0 (4,16,256,512)
  gpool(g_in, dgo0, 16, 256, 512, 128, GP0, g1, P0, gvals0, gind0); // h->(4,16,128,512)

  conv(P0, P1, DC1A, 16, 32, 128, 512);                   // (4,32,128,512)
  conv(P1, dco1, DC1B, 32, 32, 128, 512);                 // skip dco1
  {
    long long tot = (long long)B * 32 * 128 * 256;
    k_maxpool2<<<dim3((unsigned)((tot + 255) / 256)), dim3(256), 0, stream>>>(
        dco1, P0, find1, B, 32, 128, 512);                // (4,32,128,256)
  }
  gcn(g1, P0, dgo1, 32, 128, 256, 256, DG1);              // dgo1 (4,32,128,256)
  gpool(g1, dgo1, 32, 128, 256, 64, GP1, g2b, P0, gvals1, gind1);   // h->(4,32,64,256)

  // ============================ BOTTOM ===================================
  conv(P0, P1, BOT1, 32, 64, 64, 256);                    // (4,64,64,256)
  conv(P1, P0, BOT2, 64, 64, 64, 256);
  gcn(g2b, P0, P1, 64, 64, 256, 256, BOTG);               // (4,64,64,256)

  // ============================ UP PATH (u=1) ============================
  conv(P1, P0, UPC5, 64, 32, 64, 256);                    // (4,32,64,256)
  gunpool(P0, dgo1, gind1, 32, 128, 64, 256, inv1, P1);   // (4,32,128,256)
  gcn(g1, P1, P0, 32, 128, 256, 256, UG1);                // (4,32,128,256)
  unpool_concat(P0, find1, dco1, P1, 32, 128, 512);       // (4,64,128,512)
  conv(P1, P0, UPC3, 64, 32, 128, 512);
  conv(P0, P1, UPC4, 32, 32, 128, 512);

  // ============================ UP PATH (u=0) ============================
  conv(P1, P0, UPC2, 32, 16, 128, 512);                   // (4,16,128,512)
  gunpool(P0, dgo0, gind0, 16, 256, 128, 512, inv0, P1);  // (4,16,256,512)
  gcn(g_in, P1, P0, 16, 256, 512, 256, UG0);              // (4,16,256,512)
  unpool_concat(P0, find0, dco0, P1, 16, 256, 1024);      // (4,32,256,1024)
  conv(P1, P0, UPC0, 32, 16, 256, 1024);
  conv(P0, P1, UPC1, 16, 16, 256, 1024);

  // ============================ OUTPUTS ==================================
  float* outp = (float*)d_out;
  conv(P1, outp, LAST, 16, 5, 256, 1024);                 // h_final (4,5,256,1024)
  const long long OFF_SAL  = 5242880;                     // salient_spacial (4,64,64)
  const long long OFF_NODE = OFF_SAL + 16384;             // salient_node (4,64)
  const long long OFF_TS   = OFF_NODE + 256;              // ts (4,256,256)
  k_copyf<<<dim3(64), dim3(256), 0, stream>>>(g2b, outp + OFF_SAL, 16384);
  k_i2f<<<dim3(1), dim3(256), 0, stream>>>(gind1, outp + OFF_NODE, 256);
  k_copyf<<<dim3(1024), dim3(256), 0, stream>>>(g_in, outp + OFF_TS, 262144);
}
